// GAT_23003844838069
// MI455X (gfx1250) — matrix operand order repaired and checked by reading, not compiled, not run on hardware
//
#include <hip/hip_runtime.h>

typedef __attribute__((ext_vector_type(16))) _Float16 v16h;
typedef __attribute__((ext_vector_type(8)))  _Float16 v8h;
typedef __attribute__((ext_vector_type(8)))  float    v8f;

#define NEG_SLOPE 0.2f

// ---- order-preserving float <-> uint key (for atomic segment max) ----
__device__ __forceinline__ unsigned fkey(float f) {
  unsigned u = __float_as_uint(f);
  return u ^ (unsigned)(((int)u >> 31) | 0x80000000);
}
__device__ __forceinline__ float fval(unsigned k) {
  unsigned u = (k & 0x80000000u) ? (k ^ 0x80000000u) : ~k;
  return __uint_as_float(u);
}

__device__ __forceinline__ v16h cat8(v8h a, v8h b) {
  return __builtin_shufflevector(a, b, 0, 1, 2, 3, 4, 5, 6, 7,
                                       8, 9, 10, 11, 12, 13, 14, 15);
}

// =====================================================================
// WMMA GEMM, LDS-free. A fragments and B fragments are loaded as aligned
// 16-byte global loads straight in the gfx1250 VGPR fragment layout:
//   A (16-bit, 16x32): lane(half,l16) row = l16;  K = {8h..8h+7, 16+8h..16+8h+7}
//   B (16-bit, 32x16): lane(half,l16) col = l16;  K = {8h..8h+7, 16+8h..16+8h+7}
//   D (f32, 16x16):    element r -> row r+8*half, col l16
// Ah: M x 128 f16 row-major.  Bt: NCOLS x 128 f16 (W transposed, K contig).
// Block = 256 thr = 8 waves; block tile 128 x NCOLS; wave w rows [16w,16w+16).
// =====================================================================
template<int NCOLS>
__global__ __launch_bounds__(256) void gat_gemm_wmma(
    const _Float16* __restrict__ Ah, const _Float16* __restrict__ Bt,
    float* __restrict__ C, int M) {
  constexpr int K  = 128;
  constexpr int NT = NCOLS / 16;

  const int tid  = threadIdx.x;
  const int wave = tid >> 5;
  const int lane = tid & 31;
  const int half = lane >> 4;
  const int l16  = lane & 15;

  const int rowA  = blockIdx.x * 128 + wave * 16 + l16;
  const int rowAc = (rowA < M) ? rowA : 0;          // clamp OOB loads to row 0
  const _Float16* ap = Ah + (size_t)rowAc * K;

  v8f acc[NT];
  #pragma unroll
  for (int t = 0; t < NT; ++t)
    #pragma unroll
    for (int j = 0; j < 8; ++j) acc[t][j] = 0.0f;

  #pragma unroll
  for (int k0 = 0; k0 < K; k0 += 32) {
    v8h a0 = *(const v8h*)(ap + k0 + 8 * half);
    v8h a1 = *(const v8h*)(ap + k0 + 16 + 8 * half);
    v16h af = cat8(a0, a1);
    #pragma unroll
    for (int t = 0; t < NT; ++t) {
      const _Float16* bp = Bt + (size_t)(t * 16 + l16) * K + k0 + 8 * half;
      v16h bf = cat8(*(const v8h*)bp, *(const v8h*)(bp + 16));
      acc[t] = __builtin_amdgcn_wmma_f32_16x16x32_f16(
          false, af, false, bf, (short)0, acc[t], false, false);
    }
  }

  const int mBase = blockIdx.x * 128 + wave * 16 + 8 * half;
  #pragma unroll
  for (int t = 0; t < NT; ++t)
    #pragma unroll
    for (int r = 0; r < 8; ++r) {
      int m = mBase + r;
      if (m < M) C[(size_t)m * NCOLS + t * 16 + l16] = acc[t][r];
    }
}

// ---- f32 -> f16 elementwise ----
__global__ void gat_cvt_f16(const float* __restrict__ in,
                            _Float16* __restrict__ out, long long n) {
  long long i = (long long)blockIdx.x * blockDim.x + threadIdx.x;
  if (i < n) out[i] = (_Float16)in[i];
}

// ---- W [128 x NC] f32 -> Wt [NC x 128] f16 (K contiguous) ----
template<int NC>
__global__ void gat_cvt_w_t(const float* __restrict__ W,
                            _Float16* __restrict__ Wt) {
  int i = blockIdx.x * blockDim.x + threadIdx.x;
  if (i >= NC * 128) return;
  int n = i >> 7, k = i & 127;
  Wt[n * 128 + k] = (_Float16)W[k * NC + n];
}

// ---- per-node attention coefficients: a = <h[n,hd,:], att[hd,:]> ----
template<int H, int C>
__global__ void gat_attn_coef(const float* __restrict__ h,
                              const float* __restrict__ att_src,
                              const float* __restrict__ att_dst,
                              float* __restrict__ a_src,
                              float* __restrict__ a_dst, int N) {
  int i = blockIdx.x * blockDim.x + threadIdx.x;
  if (i >= N * H) return;
  int hd = i % H;
  const float* hp  = h + (size_t)i * C;        // (n*H + hd)*C == n*H*C + hd*C
  const float* asp = att_src + hd * C;
  const float* adp = att_dst + hd * C;
  float s = 0.f, d = 0.f;
  #pragma unroll
  for (int c = 0; c < C; ++c) {
    float v = hp[c];
    s += v * asp[c];
    d += v * adp[c];
  }
  a_src[i] = s;
  a_dst[i] = d;
}

__device__ __forceinline__ void edge_sd(const long long* __restrict__ src,
                                        const long long* __restrict__ dst,
                                        long long e, long long E, int& s, int& d) {
  if (e < E) { s = (int)src[e]; d = (int)dst[e]; }
  else       { s = d = (int)(e - E); }
}

// ---- pass 1: segment max of leaky(e) over dst ----
template<int H>
__global__ void gat_edge_max(const long long* __restrict__ src,
                             const long long* __restrict__ dst,
                             const float* __restrict__ a_src,
                             const float* __restrict__ a_dst,
                             unsigned* __restrict__ mkey,
                             long long E, int N) {
  long long i = (long long)blockIdx.x * blockDim.x + threadIdx.x;
  if (i >= (E + N) * H) return;
  long long e = i / H; int hd = (int)(i % H);
  int s, d; edge_sd(src, dst, e, E, s, d);
  float ev = a_src[(size_t)s * H + hd] + a_dst[(size_t)d * H + hd];
  ev = (ev > 0.f) ? ev : NEG_SLOPE * ev;
  atomicMax(&mkey[(size_t)d * H + hd], fkey(ev));
}

// ---- pass 2: segment sum of exp(e - m) over dst ----
template<int H>
__global__ void gat_edge_sum(const long long* __restrict__ src,
                             const long long* __restrict__ dst,
                             const float* __restrict__ a_src,
                             const float* __restrict__ a_dst,
                             const unsigned* __restrict__ mkey,
                             float* __restrict__ ssum,
                             long long E, int N) {
  long long i = (long long)blockIdx.x * blockDim.x + threadIdx.x;
  if (i >= (E + N) * H) return;
  long long e = i / H; int hd = (int)(i % H);
  int s, d; edge_sd(src, dst, e, E, s, d);
  float ev = a_src[(size_t)s * H + hd] + a_dst[(size_t)d * H + hd];
  ev = (ev > 0.f) ? ev : NEG_SLOPE * ev;
  float m = fval(mkey[(size_t)d * H + hd]);
  atomicAdd(&ssum[(size_t)d * H + hd], __expf(ev - m));
}

// ---- pass 3: out[dst] += alpha * h[src]  (one thread per edge-channel) ----
template<int H, int C>
__global__ void gat_edge_agg(const long long* __restrict__ src,
                             const long long* __restrict__ dst,
                             const float* __restrict__ a_src,
                             const float* __restrict__ a_dst,
                             const unsigned* __restrict__ mkey,
                             const float* __restrict__ ssum,
                             const float* __restrict__ h,
                             float* __restrict__ out,
                             long long E, int N) {
  constexpr int HC = H * C;
  long long i = (long long)blockIdx.x * blockDim.x + threadIdx.x;
  if (i >= (E + N) * HC) return;
  long long e = i / HC; int hc = (int)(i % HC); int hd = hc / C;
  int s, d; edge_sd(src, dst, e, E, s, d);
  float ev = a_src[(size_t)s * H + hd] + a_dst[(size_t)d * H + hd];
  ev = (ev > 0.f) ? ev : NEG_SLOPE * ev;
  float m  = fval(mkey[(size_t)d * H + hd]);
  float sm = ssum[(size_t)d * H + hd];
  float alpha = __expf(ev - m) / (sm + 1e-16f);
  atomicAdd(&out[(size_t)d * HC + hc], alpha * h[(size_t)s * HC + hc]);
}

// ---- bias + relu, emit f16 input for next GEMM ----
template<int STRIDE>
__global__ void gat_bias_relu_cvt(const float* __restrict__ o,
                                  const float* __restrict__ b,
                                  _Float16* __restrict__ outh, long long n) {
  long long i = (long long)blockIdx.x * blockDim.x + threadIdx.x;
  if (i >= n) return;
  float v = o[i] + b[i & (STRIDE - 1)];
  outh[i] = (_Float16)((v > 0.f) ? v : 0.f);
}

// ---- bias into final output ----
template<int STRIDE>
__global__ void gat_bias_out(const float* __restrict__ o,
                             const float* __restrict__ b,
                             float* __restrict__ out, long long n) {
  long long i = (long long)blockIdx.x * blockDim.x + threadIdx.x;
  if (i >= n) return;
  out[i] = o[i] + b[i & (STRIDE - 1)];
}

extern "C" void kernel_launch(void* const* d_in, const int* in_sizes, int n_in,
                              void* d_out, int out_size, void* d_ws, size_t ws_size,
                              hipStream_t stream) {
  const float*     x        = (const float*)d_in[0];
  const long long* ei       = (const long long*)d_in[1];  // int64 [2, E]
  const float*     W1       = (const float*)d_in[2];
  const float*     att_src1 = (const float*)d_in[3];
  const float*     att_dst1 = (const float*)d_in[4];
  const float*     b1       = (const float*)d_in[5];
  const float*     W2       = (const float*)d_in[6];
  const float*     att_src2 = (const float*)d_in[7];
  const float*     att_dst2 = (const float*)d_in[8];
  const float*     b2       = (const float*)d_in[9];

  const int       N = in_sizes[0] / 128;       // 50000
  const long long E = in_sizes[1] / 2;         // 1600000
  const long long Etot = E + N;                // with self loops
  const long long* src = ei;
  const long long* dst = ei + E;

  // workspace layout
  char* ws = (char*)d_ws;
  size_t off = 0;
  auto alloc = [&](size_t bytes) -> void* {
    void* p = ws + off;
    off = (off + bytes + 255) & ~(size_t)255;
    return p;
  };
  _Float16* xh1 = (_Float16*)alloc((size_t)N * 128 * 2);  // f16(x)
  _Float16* xh2 = (_Float16*)alloc((size_t)N * 128 * 2);  // f16(relu(o1+b1))
  _Float16* Wt1 = (_Float16*)alloc((size_t)128 * 128 * 2);
  _Float16* Wt2 = (_Float16*)alloc((size_t)64 * 128 * 2);
  float* h1  = (float*)alloc((size_t)N * 128 * 4);
  float* h2  = (float*)alloc((size_t)N * 64 * 4);
  float* as1 = (float*)alloc((size_t)N * 4 * 4);
  float* ad1 = (float*)alloc((size_t)N * 4 * 4);
  float* as2 = (float*)alloc((size_t)N * 4);
  float* ad2 = (float*)alloc((size_t)N * 4);
  // ---- contiguous zero-initialized region ----
  size_t zoff = off;
  float*    o1 = (float*)alloc((size_t)N * 128 * 4);
  float*    o2 = (float*)alloc((size_t)N * 64 * 4);
  unsigned* m1 = (unsigned*)alloc((size_t)N * 4 * 4);
  float*    s1 = (float*)alloc((size_t)N * 4 * 4);
  unsigned* m2 = (unsigned*)alloc((size_t)N * 4);
  float*    s2 = (float*)alloc((size_t)N * 4);
  size_t zbytes = off - zoff;
  hipMemsetAsync(ws + zoff, 0, zbytes, stream);  // key 0 < fkey(x) for all finite x

  const dim3 blk(256);
  const unsigned gRows = (unsigned)((N + 127) / 128);
  auto g1 = [&](long long n) { return dim3((unsigned)((n + 255) / 256)); };

  // ---- precision staging (one-time conversions, ~40MB traffic = noise) ----
  gat_cvt_f16<<<g1((long long)N * 128), blk, 0, stream>>>(x, xh1, (long long)N * 128);
  gat_cvt_w_t<128><<<g1(128 * 128), blk, 0, stream>>>(W1, Wt1);
  gat_cvt_w_t<64><<<g1(64 * 128), blk, 0, stream>>>(W2, Wt2);

  // ---------------- layer 1 (H=4, C=32, concat) ----------------
  gat_gemm_wmma<128><<<dim3(gRows), blk, 0, stream>>>(xh1, Wt1, h1, N);
  gat_attn_coef<4, 32><<<g1((long long)N * 4), blk, 0, stream>>>(h1, att_src1,
                                                                 att_dst1, as1, ad1, N);
  gat_edge_max<4><<<g1(Etot * 4), blk, 0, stream>>>(src, dst, as1, ad1, m1, E, N);
  gat_edge_sum<4><<<g1(Etot * 4), blk, 0, stream>>>(src, dst, as1, ad1, m1, s1, E, N);
  gat_edge_agg<4, 32><<<g1(Etot * 128), blk, 0, stream>>>(src, dst, as1, ad1, m1, s1,
                                                          h1, o1, E, N);
  gat_bias_relu_cvt<128><<<g1((long long)N * 128), blk, 0, stream>>>(
      o1, b1, xh2, (long long)N * 128);

  // ---------------- layer 2 (H=1, C=64, concat) ----------------
  gat_gemm_wmma<64><<<dim3(gRows), blk, 0, stream>>>(xh2, Wt2, h2, N);
  gat_attn_coef<1, 64><<<g1((long long)N), blk, 0, stream>>>(h2, att_src2,
                                                             att_dst2, as2, ad2, N);
  gat_edge_max<1><<<g1(Etot), blk, 0, stream>>>(src, dst, as2, ad2, m2, E, N);
  gat_edge_sum<1><<<g1(Etot), blk, 0, stream>>>(src, dst, as2, ad2, m2, s2, E, N);
  gat_edge_agg<1, 64><<<g1(Etot * 64), blk, 0, stream>>>(src, dst, as2, ad2, m2, s2,
                                                         h2, o2, E, N);
  gat_bias_out<64><<<g1((long long)N * 64), blk, 0, stream>>>(o2, b2, (float*)d_out,
                                                              (long long)N * 64);
}
